// WDRCModule_42047729828204
// MI455X (gfx1250) — matrix operand order repaired and checked by reading, not compiled, not run on hardware
//
#include <hip/hip_runtime.h>

typedef __attribute__((ext_vector_type(16))) __bf16 v16bf;
typedef __attribute__((ext_vector_type(8)))  float  v8f;
typedef __attribute__((ext_vector_type(4)))  int    v4i;

#ifndef __has_builtin
#define __has_builtin(x) 0
#endif
#if __has_builtin(__builtin_amdgcn_global_load_async_to_lds_b128)
#define HAVE_ASYNC_LDS 1
#else
#define HAVE_ASYNC_LDS 0
#endif
#define AS1 __attribute__((address_space(1)))
#define AS3 __attribute__((address_space(3)))

namespace {
constexpr int NB  = 16;             // batch
constexpr int NT  = 6000;           // time frames
constexpr int NF  = 257;            // freq bins
constexpr int NM  = 64;             // mel bands
constexpr int NR  = NB * NT;        // 96000 GEMM rows
constexpr int FPAD_K = 288;         // freq padded to mult of 32 (GEMM1 K)
constexpr int FPAD_N = 272;         // freq padded to mult of 16 (GEMM2 N)
constexpr int FTILES = FPAD_N / 16; // 17
constexpr int MELFB_LSTRIDE = 296;  // LDS row stride (bank-conflict-free)
constexpr int FBNT_LSTRIDE  = 72;   // LDS row stride (bank-conflict-free)
}

union V16 { uint4 u[2]; v16bf v; };

__device__ __forceinline__ float fast_log2(float v) { return __builtin_amdgcn_logf(v); }
__device__ __forceinline__ float fast_exp2(float v) { return __builtin_amdgcn_exp2f(v); }

// Cooperative stage of a [NROWS x ROWLEN] bf16 panel into LDS with padded row
// stride, via gfx1250 async global->LDS b128 copies when available.
template <int NROWS, int ROWLEN, int ROWSTRIDE>
__device__ __forceinline__ void stage_to_lds(const __bf16* __restrict__ g, __bf16* l)
{
  constexpr int CHUNKS = ROWLEN / 8;           // 16B chunks per row
  constexpr int TOTAL  = NROWS * CHUNKS;
  for (int c = threadIdx.x; c < TOTAL; c += 256) {
    const int r = c / CHUNKS;
    const int k = (c - r * CHUNKS) * 8;
#if HAVE_ASYNC_LDS
    __builtin_amdgcn_global_load_async_to_lds_b128(
        (AS1 v4i*)(void*)(g + (size_t)r * ROWLEN + k),
        (AS3 v4i*)(l + (size_t)r * ROWSTRIDE + k), 0, 0);
#else
    *(uint4*)(l + (size_t)r * ROWSTRIDE + k) =
        *(const uint4*)(g + (size_t)r * ROWLEN + k);
#endif
  }
}

__device__ __forceinline__ void stage_fence()
{
#if HAVE_ASYNC_LDS
# if __has_builtin(__builtin_amdgcn_s_wait_asynccnt)
  __builtin_amdgcn_s_wait_asynccnt(0);
# else
  asm volatile("s_wait_asynccnt 0" ::: "memory");
# endif
#endif
  __syncthreads();
}

// ---------------------------------------------------------------------------
// Prep: mel_fb (f32 [64x257]) -> bf16 [64x288] (zero-padded K) and
//       column-normalized transposed fb_norm^T bf16 [272x64]
// ---------------------------------------------------------------------------
__global__ void k_prep(const float* __restrict__ mel_fb,
                       __bf16* __restrict__ melfb,
                       __bf16* __restrict__ fbnT)
{
  const int f = blockIdx.x * blockDim.x + threadIdx.x;
  if (f >= FPAD_K) return;
  if (f < NF) {
    float s = 0.0f;
    for (int m = 0; m < NM; ++m) s += mel_fb[(size_t)m * NF + f];
    const float inv = 1.0f / (s + 1e-8f);
    for (int m = 0; m < NM; ++m) {
      const float v = mel_fb[(size_t)m * NF + f];
      melfb[(size_t)m * FPAD_K + f] = (__bf16)v;
      fbnT[(size_t)f * NM + m]      = (__bf16)(v * inv);
    }
  } else {
    for (int m = 0; m < NM; ++m) melfb[(size_t)m * FPAD_K + f] = (__bf16)0.0f;
    if (f < FPAD_N)
      for (int m = 0; m < NM; ++m) fbnT[(size_t)f * NM + m] = (__bf16)0.0f;
  }
}

// ---------------------------------------------------------------------------
// GEMM1 + dB: xdb[r,m] = 10*log10( sum_f (x[r,f]^2 * mel_fb[m,f]) + eps )
// B panel (all 64 mel filters) staged in LDS; one wave per 16-row tile.
// ---------------------------------------------------------------------------
__global__ __launch_bounds__(256) void k_gemm1_db(
    const float* __restrict__ x, const __bf16* __restrict__ melfb,
    float* __restrict__ xdb)
{
  __shared__ __bf16 sB[NM * MELFB_LSTRIDE];          // 37,888 B
  stage_to_lds<NM, FPAD_K, MELFB_LSTRIDE>(melfb, sB);
  stage_fence();

  const int wave = blockIdx.x * 8 + (threadIdx.x >> 5);
  const int lane = threadIdx.x & 31;
  const int lr   = lane & 15;
  const int hs   = lane >> 4;          // half-wave select (ISA K split)
  const int row0 = wave * 16;
  const float* __restrict__ xrow = x + (size_t)(row0 + lr) * NF;

  v8f acc0 = {}, acc1 = {}, acc2 = {}, acc3 = {};

  for (int k0 = 0; k0 < FPAD_K; k0 += 32) {
    // A tile (16x32 bf16): lane<16 holds K=k0..k0+15; lane>=16 holds K=k0+16..k0+31
    v16bf a;
    const int kb = k0 + hs * 16;
    if (k0 + 31 < NF) {
      #pragma unroll
      for (int j = 0; j < 8; ++j) {
        const float v1 = xrow[kb + j];
        const float v2 = xrow[kb + 8 + j];
        a[j]     = (__bf16)(v1 * v1);
        a[j + 8] = (__bf16)(v2 * v2);
      }
    } else {  // last K chunk: only f==256 is live
      #pragma unroll
      for (int j = 0; j < 8; ++j) {
        const int f1 = kb + j, f2 = kb + 8 + j;
        const float v1 = (f1 < NF) ? xrow[f1] : 0.0f;
        const float v2 = (f2 < NF) ? xrow[f2] : 0.0f;
        a[j]     = (__bf16)(v1 * v1);
        a[j + 8] = (__bf16)(v2 * v2);
      }
    }
    // B tiles from LDS (element (K=f, N=m) = melfb[m][f]; contiguous in K)
    const int bo = k0 + hs * 16;
    V16 b0, b1, b2, b3;
    {
      const __bf16* bp = &sB[(0 * 16 + lr) * MELFB_LSTRIDE + bo];
      b0.u[0] = *(const uint4*)(bp);  b0.u[1] = *(const uint4*)(bp + 8);
    }
    {
      const __bf16* bp = &sB[(1 * 16 + lr) * MELFB_LSTRIDE + bo];
      b1.u[0] = *(const uint4*)(bp);  b1.u[1] = *(const uint4*)(bp + 8);
    }
    {
      const __bf16* bp = &sB[(2 * 16 + lr) * MELFB_LSTRIDE + bo];
      b2.u[0] = *(const uint4*)(bp);  b2.u[1] = *(const uint4*)(bp + 8);
    }
    {
      const __bf16* bp = &sB[(3 * 16 + lr) * MELFB_LSTRIDE + bo];
      b3.u[0] = *(const uint4*)(bp);  b3.u[1] = *(const uint4*)(bp + 8);
    }
    acc0 = __builtin_amdgcn_wmma_f32_16x16x32_bf16(false, a, false, b0.v, (short)0, acc0, false, false);
    acc1 = __builtin_amdgcn_wmma_f32_16x16x32_bf16(false, a, false, b1.v, (short)0, acc1, false, false);
    acc2 = __builtin_amdgcn_wmma_f32_16x16x32_bf16(false, a, false, b2.v, (short)0, acc2, false, false);
    acc3 = __builtin_amdgcn_wmma_f32_16x16x32_bf16(false, a, false, b3.v, (short)0, acc3, false, false);
  }

  // C layout: VGPR i -> row = row0 + hs*8 + i ; N = lr
  #pragma unroll
  for (int i = 0; i < 8; ++i) {
    const int row = row0 + hs * 8 + i;
    float* __restrict__ p = xdb + (size_t)row * NM + lr;
    p[0]  = 3.0102999566f * fast_log2(acc0[i] + 1e-8f);  // 10*log10
    p[16] = 3.0102999566f * fast_log2(acc1[i] + 1e-8f);
    p[32] = 3.0102999566f * fast_log2(acc2[i] + 1e-8f);
    p[48] = 3.0102999566f * fast_log2(acc3[i] + 1e-8f);
  }
}

// ---------------------------------------------------------------------------
// AREMA scan (serial over T) + soft-knee compressor + limiter -> gain_lin bf16
// ---------------------------------------------------------------------------
__global__ __launch_bounds__(256) void k_scan(const float* __restrict__ xdb,
                                              __bf16* __restrict__ gain)
{
  const int c = blockIdx.x * blockDim.x + threadIdx.x;  // 0..1023
  const int b = c >> 6, m = c & 63;
  const float* __restrict__ src = xdb + (size_t)b * NT * NM + m;
  __bf16* __restrict__ dst      = gain + (size_t)b * NT * NM + m;

  const float a_c = 0.95923781f;   // 1 - exp(-0.016/0.005)
  const float r_c = 0.27385095f;   // 1 - exp(-0.016/0.05)
  const float slope = ((45.0f + 5.0f / 3.0f) - 50.0f) * 0.1f;

  float y = 0.0f;
  for (int t = 0; t < NT; ++t) {
    if ((t & 31) == 0)
      __builtin_prefetch(src + (size_t)(t + 96) * NM, 0, 1);  // global_prefetch_b8
    const float xv = src[(size_t)t * NM];
    const float al = (xv - y > 0.0f) ? a_c : r_c;
    y = y + al * (xv - y);
    const float xG = y;
    const float low  = (40.0f - xG > 0.0f) ? 1.0f : 0.0f;
    const float comp = (xG - 50.0f > 0.0f) ? 1.0f : 0.0f;
    const float knee = 1.0f - low - comp;
    const float yG = (xG + 10.0f) * low
                   + (45.0f + (xG - 45.0f) * (1.0f / 3.0f)) * comp
                   + (50.0f + slope * (xG - 40.0f)) * knee;
    const float lim = (2.0f * (yG - 90.0f) - 10.0f > 0.0f) ? 1.0f : 0.0f;
    float yL = 90.0f * lim + yG * (1.0f - lim);
    const float tr = (10.0f - 2.0f * fabsf(yG - 90.0f) > 0.0f) ? 1.0f : 0.0f;
    const float d  = yG - 85.0f;
    yL = (yG - d * d * 0.05f) * tr + yL * (1.0f - tr);
    const float gl = fast_exp2((yL - xG) * 0.16609640f);  // 10^(gdb/20)
    dst[(size_t)t * NM] = (__bf16)gl;
  }
}

// ---------------------------------------------------------------------------
// GEMM2 + apply: out[r,f] = x[r,f] * sum_m gain[r,m] * fb_norm[m,f]
// One wave per 16-row tile; A (gain) held in regs, B from LDS, loop f-tiles.
// ---------------------------------------------------------------------------
__global__ __launch_bounds__(256) void k_gemm2_apply(
    const float* __restrict__ x, const __bf16* __restrict__ gain,
    const __bf16* __restrict__ fbnT, float* __restrict__ out)
{
  __shared__ __bf16 sB[FPAD_N * FBNT_LSTRIDE];       // 39,168 B
  stage_to_lds<FPAD_N, NM, FBNT_LSTRIDE>(fbnT, sB);
  stage_fence();

  const int wave = blockIdx.x * 8 + (threadIdx.x >> 5);
  const int lane = threadIdx.x & 31;
  const int lr = lane & 15, hs = lane >> 4;
  const int row0 = wave * 16;

  // A tiles (gain rows) in registers, reused for all 17 f-tiles
  const __bf16* __restrict__ arow = gain + (size_t)(row0 + lr) * NM;
  V16 a0, a1;
  a0.u[0] = *(const uint4*)(arow + hs * 8);
  a0.u[1] = *(const uint4*)(arow + hs * 8 + 16);
  a1.u[0] = *(const uint4*)(arow + 32 + hs * 8);
  a1.u[1] = *(const uint4*)(arow + 32 + hs * 8 + 16);

  for (int ft = 0; ft < FTILES; ++ft) {
    V16 b0, b1;
    const __bf16* bp = &sB[(ft * 16 + lr) * FBNT_LSTRIDE + hs * 8];
    b0.u[0] = *(const uint4*)(bp);
    b0.u[1] = *(const uint4*)(bp + 16);
    b1.u[0] = *(const uint4*)(bp + 32);
    b1.u[1] = *(const uint4*)(bp + 48);

    v8f acc = {};
    acc = __builtin_amdgcn_wmma_f32_16x16x32_bf16(false, a0.v, false, b0.v, (short)0, acc, false, false);
    acc = __builtin_amdgcn_wmma_f32_16x16x32_bf16(false, a1.v, false, b1.v, (short)0, acc, false, false);

    const int fcol = ft * 16 + lr;
    if (fcol < NF) {          // divergence only after the WMMAs
      #pragma unroll
      for (int i = 0; i < 8; ++i) {
        const int row = row0 + hs * 8 + i;
        const size_t idx = (size_t)row * NF + fcol;
        out[idx] = x[idx] * acc[i];
      }
    }
  }
}

// ---------------------------------------------------------------------------
extern "C" void kernel_launch(void* const* d_in, const int* in_sizes, int n_in,
                              void* d_out, int out_size, void* d_ws, size_t ws_size,
                              hipStream_t stream)
{
  (void)in_sizes; (void)n_in; (void)out_size; (void)ws_size;
  const float* x      = (const float*)d_in[0];
  const float* mel_fb = (const float*)d_in[1];
  float* out = (float*)d_out;

  char* ws = (char*)d_ws;
  __bf16* gain  = (__bf16*)ws;                                      // 96000*64 bf16
  __bf16* melfb = (__bf16*)(ws + (size_t)NR * NM * sizeof(__bf16)); // 64*288 bf16
  __bf16* fbnT  = melfb + (size_t)NM * FPAD_K;                      // 272*64 bf16
  float*  xdb   = out;  // reuse d_out as f32 scratch (fully overwritten later)

  k_prep<<<3, 96, 0, stream>>>(mel_fb, melfb, fbnT);
  k_gemm1_db<<<NR / 16 / 8, 256, 0, stream>>>(x, melfb, xdb);
  k_scan<<<(NB * NM) / 256, 256, 0, stream>>>(xdb, gain);
  k_gemm2_apply<<<NR / 16 / 8, 256, 0, stream>>>(x, gain, fbnT, out);
}
